// MultiheadAttention_42571715838605
// MI455X (gfx1250) — matrix operand order repaired and checked by reading, not compiled, not run on hardware
//
#include <hip/hip_runtime.h>

typedef __attribute__((ext_vector_type(16))) _Float16     v16h;
typedef __attribute__((ext_vector_type(8)))  _Float16     v8h;
typedef __attribute__((ext_vector_type(8)))  float        v8f;
typedef __attribute__((ext_vector_type(4)))  unsigned int v4u;
typedef __attribute__((ext_vector_type(8)))  int          v8i;
typedef __attribute__((ext_vector_type(4)))  int          v4i;

#define S_LEN   2048
#define NHEADS  16
#define HD      64
#define DIM     1024
#define C3      3072                 // 3*DIM
#define QTILE   128                  // q rows per block (8 waves * 16)
#define KTILE   64                   // keys per iteration
#define NTILES  (S_LEN / KTILE)      // 32
#define KF_STR  68                   // f32 row stride in staging buffers (64+4 pad)
#define KS_STR  72                   // f16 row stride, K tile  (64 + 8 pad)
#define VT_STR  72                   // f16 row stride, V^T tile (64 + 8 pad)
#define OFF32   (32 * KF_STR * 4)    // byte offset of key row +32 in f32 buffers

// One wave-wide async DMA: each lane copies 16B global -> LDS (ASYNCcnt-tracked)
__device__ __forceinline__ void async_b128(unsigned int lds_off, const float* g) {
    asm volatile("global_load_async_to_lds_b128 %0, %1, off"
                 :: "v"(lds_off), "v"((unsigned long long)(uintptr_t)g)
                 : "memory");
}

// TDM: DMA one 64-row x 64-f32 tile (row stride C3 elements) into LDS with
// 4-DWORD padding after every 64 DWORDs (-> KF_STR=68 rows). TENSORcnt-tracked.
__device__ __forceinline__ void tdm_load_k(const float* g, unsigned int lds_off) {
    const unsigned long long ga = (unsigned long long)(uintptr_t)g;
    v4u g0;
    g0[0] = 1u;                                          // count=1 (valid user D#)
    g0[1] = lds_off;                                     // lds_addr[63:32]
    g0[2] = (unsigned int)ga;                            // global_addr[95:64]
    g0[3] = (unsigned int)(ga >> 32) | (2u << 30);       // addr[120:96] | type=2
    v8i g1;
    g1[0] = (int)((2u << 16) | (1u << 20) | (5u << 22) | (3u << 25));
        // data_size=4B | pad_enable | pad_interval=64dw | pad_amount=4dw
    g1[1] = (int)(64u << 16);                            // tensor_dim0[79:48] = 64
    g1[2] = (int)(2048u << 16);                          // tensor_dim1[111:80] = 2048
    g1[3] = (int)(64u << 16);                            // tile_dim0[127:112] = 64
    g1[4] = 64;                                          // tile_dim1[143:128] = 64
    g1[5] = C3;                                          // tensor_dim0_stride[207:160]
    g1[6] = 0;
    g1[7] = 0;
    v4i gz4 = {0, 0, 0, 0};                              // 2D tensor: groups 2/3 unused
    v8i gz8 = {0, 0, 0, 0, 0, 0, 0, 0};
    __builtin_amdgcn_tensor_load_to_lds(g0, g1, gz4, gz4, gz8, 0);
}

__global__ __launch_bounds__(256)
void fa_fwd_kernel(const float* __restrict__ x, float* __restrict__ out) {
    __shared__ __align__(16) float    Kf[2][KTILE * KF_STR];   // TDM K landing (f32)
    __shared__ __align__(16) float    Vf[2][KTILE * KF_STR];   // async V landing (f32)
    __shared__ __align__(16) _Float16 Ks[KTILE * KS_STR];      // K tile f16 [key][d]
    __shared__ __align__(16) _Float16 VT[HD * VT_STR];         // V^T f16 [d][key]

    const int tid  = threadIdx.x;
    const int wave = tid >> 5;
    const int lane = tid & 31;
    const int hf   = lane >> 4;      // lane-group: selects K-subrange of fragments
    const int n    = lane & 15;      // fragment column / row index

    const int qt = blockIdx.x;       // q tile
    const int h  = blockIdx.y;       // head
    const int b  = blockIdx.z;       // batch
    const int q0 = qt * QTILE + wave * 16;

    // 1/sqrt(64) * log2(e): softmax in base-2 (raw v_exp_f32 is base-2)
    const float scale = 0.125f * 1.44269504088896340736f;

    // ---- Q tile -> two B fragments of Q^T (lane = column q = n,
    //      halves = contiguous d range 32c + 16*hf + e), scaled ----
    v16h bq0{}, bq1{};
    {
        const float* qp = x + (size_t)(b * S_LEN + q0 + n) * C3 + h * 192 + hf * 16;
#pragma unroll
        for (int e = 0; e < 16; ++e) {
            bq0[e] = (_Float16)(qp[e]      * scale);
            bq1[e] = (_Float16)(qp[32 + e] * scale);
        }
    }

    v8f o0 = {}, o1 = {}, o2 = {}, o3 = {};   // O accumulators (4 d-chunks)
    float mcur = -1e30f, lcur = 0.0f;         // online-softmax state for q = n

    // staging split (K convert + V async): thread -> (key rows skey, skey+32; seg)
    const int skey = tid >> 3;
    const int sseg = tid & 7;
    const float* kbase = x + (size_t)b * S_LEN * C3 + (size_t)skey * C3
                           + h * 192 + 64 + sseg * 8;
    // K tile global base for the TDM (row 0 of the (b,h) K matrix)
    const float* ktens = x + (size_t)b * S_LEN * C3 + h * 192 + 64;

    unsigned int vfo[2], kfl[2];
    vfo[0] = (unsigned int)(uintptr_t)&Vf[0][skey * KF_STR + sseg * 8];
    vfo[1] = (unsigned int)(uintptr_t)&Vf[1][skey * KF_STR + sseg * 8];
    kfl[0] = (unsigned int)(uintptr_t)&Kf[0][0];
    kfl[1] = (unsigned int)(uintptr_t)&Kf[1][0];

    auto issue_tile = [&](int kt, int bsel) {
        if (wave == 0) tdm_load_k(ktens + (size_t)kt * C3, kfl[bsel]);
        const float* g0 = kbase + (size_t)kt * C3;            // V, key row skey
        const float* g1 = g0 + (size_t)32 * C3;               // V, key row skey+32
        async_b128(vfo[bsel],              g0 + 64);
        async_b128(vfo[bsel] + 16,         g0 + 68);
        async_b128(vfo[bsel] + OFF32,      g1 + 64);
        async_b128(vfo[bsel] + OFF32 + 16, g1 + 68);
    };

    issue_tile(0, 0);   // prologue

    for (int j = 0; j < NTILES; ++j) {
        const int cur = j & 1;
        asm volatile("s_wait_asynccnt 0" ::: "memory");       // V tile j landed
        if (wave == 0) __builtin_amdgcn_s_wait_tensorcnt(0);  // K tile j landed
        __syncthreads();                                      // visible to all waves;
                                                              // prev iter consumed
        if (j + 1 < NTILES) issue_tile((j + 1) * KTILE, cur ^ 1);

        // ---- convert pass: f32 -> f16 row-major K; f16 transposed V^T with
        //      key-pair packing (two adjacent keys -> one b32 store) ----
        {
            const float* kr = &Kf[cur][skey * KF_STR + sseg * 8];
            v8h kh0;
#pragma unroll
            for (int e = 0; e < 8; ++e) kh0[e] = (_Float16)kr[e];
            *(v8h*)&Ks[skey * KS_STR + sseg * 8] = kh0;
            const float* kr1 = kr + 32 * KF_STR;
            v8h kh1;
#pragma unroll
            for (int e = 0; e < 8; ++e) kh1[e] = (_Float16)kr1[e];
            *(v8h*)&Ks[(skey + 32) * KS_STR + sseg * 8] = kh1;

            const int kp = tid >> 3;          // key pair 0..31 -> keys 2kp, 2kp+1
            const float* va = &Vf[cur][(2 * kp) * KF_STR + sseg * 8];
            const float* vb = va + KF_STR;
#pragma unroll
            for (int e = 0; e < 8; ++e) {
                union { _Float16 hh[2]; unsigned int u; } pk;
                pk.hh[0] = (_Float16)va[e];
                pk.hh[1] = (_Float16)vb[e];
                *(unsigned int*)&VT[(sseg * 8 + e) * VT_STR + 2 * kp] = pk.u;
            }
        }
        __syncthreads();

        // ---- S^T = K * Q^T : four 16x16 C tiles (keys t*16..t*16+15), 8 WMMA ----
        v8f st[4] = {{}, {}, {}, {}};
#pragma unroll
        for (int c = 0; c < 2; ++c) {
            const v16h qb = c ? bq1 : bq0;
#pragma unroll
            for (int t = 0; t < 4; ++t) {
                // A fragment of K: lane = key row, halves = d in 16hf+0..15
                v16h a;
                const _Float16* p = &Ks[(t * 16 + n) * KS_STR + c * 32 + hf * 16];
#pragma unroll
                for (int e = 0; e < 8; ++e) { a[e] = p[e]; a[e + 8] = p[8 + e]; }
                st[t] = __builtin_amdgcn_wmma_f32_16x16x32_f16(false, a, false, qb,
                                                               (short)0, st[t], false, false);
            }
        }

        // ---- online softmax: lane holds 32 logits of row q=n ----
        float mx = st[0][0];
#pragma unroll
        for (int t = 0; t < 4; ++t)
#pragma unroll
            for (int i = 0; i < 8; ++i) mx = fmaxf(mx, st[t][i]);
        mx = fmaxf(mx, __shfl_xor(mx, 16, 32));
        const float mn = fmaxf(mcur, mx);
        const float al = __builtin_amdgcn_exp2f(mcur - mn);  // raw v_exp_f32

        // exp results feed the two P A-fragments directly (S^T C layout == A layout)
        v16h ap0, ap1;
        float rs = 0.0f;
#pragma unroll
        for (int i = 0; i < 8; ++i) {
            const float t0 = __builtin_amdgcn_exp2f(st[0][i] - mn);
            const float t1 = __builtin_amdgcn_exp2f(st[1][i] - mn);
            const float t2 = __builtin_amdgcn_exp2f(st[2][i] - mn);
            const float t3 = __builtin_amdgcn_exp2f(st[3][i] - mn);
            rs += (t0 + t1) + (t2 + t3);
            ap0[i] = (_Float16)t0;  ap0[i + 8] = (_Float16)t1;
            ap1[i] = (_Float16)t2;  ap1[i + 8] = (_Float16)t3;
        }
        rs += __shfl_xor(rs, 16, 32);
        lcur = lcur * al + rs;
        mcur = mn;

        // rescale O rows (row q = i + 8*hf needs alpha from lane i + 8*hf)
#pragma unroll
        for (int i = 0; i < 8; ++i) {
            const float ai = __shfl(al, i + 8 * hf, 32);
            o0[i] *= ai; o1[i] *= ai; o2[i] *= ai; o3[i] *= ai;
        }

        // ---- O += P * V : 8 WMMA (4 d-chunks x 2 key-chunks), B frags from V^T ----
#pragma unroll
        for (int c = 0; c < 4; ++c) {
            v16h bv0, bv1;
            const _Float16* vr = &VT[(c * 16 + n) * VT_STR + hf * 8];
#pragma unroll
            for (int e = 0; e < 8; ++e) {
                bv0[e] = vr[e]; bv0[e + 8] = vr[16 + e];
                bv1[e] = vr[32 + e]; bv1[e + 8] = vr[48 + e];
            }
            v8f& oc = (c == 0 ? o0 : c == 1 ? o1 : c == 2 ? o2 : o3);
            oc = __builtin_amdgcn_wmma_f32_16x16x32_f16(false, ap0, false, bv0,
                                                        (short)0, oc, false, false);
            oc = __builtin_amdgcn_wmma_f32_16x16x32_f16(false, ap1, false, bv1,
                                                        (short)0, oc, false, false);
        }
    }

    // ---- normalize and write out (coalesced across each 16-lane half) ----
#pragma unroll
    for (int i = 0; i < 8; ++i) {
        const float li  = __shfl(lcur, i + 8 * hf, 32);
        const float inv = 1.0f / li;
        const int qr = q0 + i + 8 * hf;
        float* orow = out + (size_t)(b * S_LEN + qr) * DIM + h * HD + n;
        orow[0]  = o0[i] * inv;
        orow[16] = o1[i] * inv;
        orow[32] = o2[i] * inv;
        orow[48] = o3[i] * inv;
    }
}

extern "C" void kernel_launch(void* const* d_in, const int* in_sizes, int n_in,
                              void* d_out, int out_size, void* d_ws, size_t ws_size,
                              hipStream_t stream) {
    (void)n_in; (void)out_size; (void)d_ws; (void)ws_size;
    const float* x = (const float*)d_in[0];
    float* out = (float*)d_out;
    const int B = in_sizes[0] / (S_LEN * C3);   // = 4
    dim3 grid(S_LEN / QTILE, NHEADS, B);
    fa_fwd_kernel<<<grid, 256, 0, stream>>>(x, out);
}
